// Loss_fun_50611894616711
// MI455X (gfx1250) — compile-verified, matrix operand order checked by reading, not run on hardware
//
#include <hip/hip_runtime.h>
#include <hip/hip_bf16.h>
#include <cstdint>
#include <cstddef>

// ---------------- problem constants (match reference) ----------------
constexpr int   NN   = 20000;      // nodes
constexpr int   DD   = 128;        // embedding dim
constexpr int   VV   = 3;          // views
constexpr int   NPOS = 512;
constexpr int   NNEG = 1024;
constexpr int   LL   = NPOS + NNEG;   // 1536 labeled
constexpr int   KS   = LL * VV;       // 4608 sup rows (288 tiles, even)
constexpr int   UU   = 2048;
constexpr int   KU   = UU * VV;       // 6144 unsup rows (384 tiles, even)
constexpr float INV_TEMP = 5.0f;      // 1/0.2
constexpr float L_UNSUP  = 0.2f;
constexpr int   BCE_BLOCKS = 64;
constexpr int   LAB_TILES  = (NPOS * VV) / 16;   // 96: tiles with label==1 rows
constexpr float CNT_POS = 2.0f + (NPOS - 1) * 3.0f;  // 1535
constexpr float CNT_NEG = 2.0f + (NNEG - 1) * 3.0f;  // 3071

// ---------------- vector types ----------------
typedef __attribute__((ext_vector_type(16))) _Float16 v16h;
typedef __attribute__((ext_vector_type(8)))  _Float16 v8h;
typedef __attribute__((ext_vector_type(4)))  _Float16 v4h;
typedef __attribute__((ext_vector_type(8)))  float    v8f;

union V16 { v16h v; v8h h[2]; };

// =====================================================================
// Kernel 1: masked BCE partial sums (fused + 3 views), per-block partials
// =====================================================================
__device__ __forceinline__ float bce_logits(float x, float y) {
    return fmaxf(x, 0.0f) - x * y + log1pf(__expf(-fabsf(x)));
}

__global__ void bce_partials(const float* __restrict__ fused,
                             const float* __restrict__ views,
                             const float* __restrict__ labels,
                             const uint8_t* __restrict__ mask,
                             float* __restrict__ part) {
    float acc[5] = {0.f, 0.f, 0.f, 0.f, 0.f};
    const int stride = gridDim.x * blockDim.x;
    for (int i = blockIdx.x * blockDim.x + threadIdx.x; i < NN; i += stride) {
        const float y  = labels[i];
        const float mf = mask[i] ? 1.0f : 0.0f;
        acc[0] += mf;
        acc[1] += bce_logits(fused[i], y) * mf;
#pragma unroll
        for (int v = 0; v < VV; ++v)
            acc[2 + v] += bce_logits(views[v * NN + i], y) * mf;
    }
#pragma unroll
    for (int q = 0; q < 5; ++q)
#pragma unroll
        for (int off = 16; off >= 1; off >>= 1)
            acc[q] += __shfl_xor(acc[q], off, 32);

    __shared__ float red[5][8];
    const int wid = threadIdx.x >> 5, lane = threadIdx.x & 31;
    if (lane == 0)
#pragma unroll
        for (int q = 0; q < 5; ++q) red[q][wid] = acc[q];
    __syncthreads();
    if (threadIdx.x == 0) {
#pragma unroll
        for (int q = 0; q < 5; ++q) {
            float s = 0.f;
            for (int w = 0; w < 8; ++w) s += red[q][w];
            part[blockIdx.x * 5 + q] = s;
        }
    }
}

// =====================================================================
// Kernel 2: gather proj rows, L2-normalize, store f16 row-major [rows][128]
// =====================================================================
__global__ void gather_normalize(const float* __restrict__ proj,
                                 const int* __restrict__ idx0,
                                 const int* __restrict__ idx1,
                                 int nPos, int rows,
                                 _Float16* __restrict__ zout) {
    const int wid  = threadIdx.x >> 5;
    const int lane = threadIdx.x & 31;
    const int row  = blockIdx.x * (blockDim.x >> 5) + wid;
    if (row >= rows) return;

    const int node = row / VV;
    const int view = row - node * VV;
    const int idx  = (node < nPos) ? idx0[node] : idx1[node - nPos];

    const float* src = proj + ((size_t)view * NN + (size_t)idx) * DD;
    const float4 v4 = reinterpret_cast<const float4*>(src)[lane];

    float ss = v4.x * v4.x + v4.y * v4.y + v4.z * v4.z + v4.w * v4.w;
#pragma unroll
    for (int off = 16; off >= 1; off >>= 1) ss += __shfl_xor(ss, off, 32);

    const float rn = 1.0f / fmaxf(sqrtf(ss), 1e-8f);
    v4h o;
    o[0] = (_Float16)(v4.x * rn);
    o[1] = (_Float16)(v4.y * rn);
    o[2] = (_Float16)(v4.z * rn);
    o[3] = (_Float16)(v4.w * rn);
    *reinterpret_cast<v4h*>(zout + (size_t)row * DD + lane * 4) = o;
}

// =====================================================================
// Kernel 3: WMMA Gram + fused row reductions (templated mode),
// double-buffered B operand; lean epilogue (no persistent index arrays,
// common path branchless via uniform same-label scalar).
//   denom_j = sum_{k != j} exp(s_jk)            (both modes)
//   MODE 0 (sup):  possim_j = sum over same-label, k!=j of s_jk
//   MODE 1 (uns):  possim_j = sum over same-node, diff-view of s_jk
// =====================================================================
__device__ __forceinline__ void load_operand(const _Float16* __restrict__ z,
                                             int rowBase, int kOff,
                                             int grp, int r, V16& u) {
    // ISA 16-bit operand layout: lane holds K-chunks [8g,8g+8) and
    // [16+8g, 24+8g) of row (rowBase + (lane&15)).
    const _Float16* p = z + (size_t)(rowBase + r) * DD + kOff;
    u.h[0] = *reinterpret_cast<const v8h*>(p + grp * 8);
    u.h[1] = *reinterpret_cast<const v8h*>(p + 16 + grp * 8);
}

__device__ __forceinline__ void load_btile(const _Float16* __restrict__ z,
                                           int kt, int grp, int r, V16* bU) {
#pragma unroll
    for (int c = 0; c < 4; ++c) load_operand(z, kt * 16, 32 * c, grp, r, bU[c]);
}

template <int MODE>
__global__ __launch_bounds__(32)
void gram_rowstats(const _Float16* __restrict__ z, int K,
                   float* __restrict__ denomO,
                   float* __restrict__ possimO) {
    const int lane = threadIdx.x;
    const int grp  = lane >> 4;
    const int r    = lane & 15;
    const int jt   = blockIdx.x;

    V16 aU[4];
#pragma unroll
    for (int c = 0; c < 4; ++c) load_operand(z, jt * 16, 32 * c, grp, r, aU[c]);

    float dacc[8], pacc[8];
#pragma unroll
    for (int v = 0; v < 8; ++v) { dacc[v] = 0.f; pacc[v] = 0.f; }

    const bool labj = (jt < LAB_TILES);   // MODE 0: row label uniform per block
    const int  ntiles = K >> 4;           // even (288 / 384)

    auto compute_tile = [&](int kt, const V16* bU) {
        v8f acc = {};
        acc = __builtin_amdgcn_wmma_f32_16x16x32_f16(false, aU[0].v, false, bU[0].v,
                                                     (short)0, acc, false, false);
        acc = __builtin_amdgcn_wmma_f32_16x16x32_f16(false, aU[1].v, false, bU[1].v,
                                                     (short)0, acc, false, false);
        acc = __builtin_amdgcn_wmma_f32_16x16x32_f16(false, aU[2].v, false, bU[2].v,
                                                     (short)0, acc, false, false);
        acc = __builtin_amdgcn_wmma_f32_16x16x32_f16(false, aU[3].v, false, bU[3].v,
                                                     (short)0, acc, false, false);

        if constexpr (MODE == 0) {
            if (kt != jt) {
                // uniform per-tile same-label scalar: pacc += s * slf
                const float slf = (labj == (kt < LAB_TILES)) ? 1.0f : 0.0f;
#pragma unroll
                for (int v = 0; v < 8; ++v) {
                    const float s = acc[v] * INV_TEMP;
                    dacc[v] += __expf(s);
                    pacc[v] = fmaf(s, slf, pacc[v]);
                }
            } else {                    // diagonal tile (rare: 1 per block)
                const int k = (kt << 4) + r;
#pragma unroll
                for (int v = 0; v < 8; ++v) {
                    const float s = acc[v] * INV_TEMP;
                    const float e = __expf(s);
                    const bool offd = ((jt << 4) + v + (grp << 3)) != k;
                    dacc[v] += offd ? e : 0.0f;
                    pacc[v] += offd ? s : 0.0f;
                }
            }
        } else {
            if (kt + 1 < jt || kt > jt + 1) {   // far tile: denom only
#pragma unroll
                for (int v = 0; v < 8; ++v)
                    dacc[v] += __expf(acc[v] * INV_TEMP);
            } else {                    // band tile (rare: <=3 per block)
                const int k  = (kt << 4) + r;
                const int ik = k / VV;
                const int mk = k - VV * ik;
#pragma unroll
                for (int v = 0; v < 8; ++v) {
                    const int jvv = (jt << 4) + v + (grp << 3);
                    const int ijv = jvv / VV;
                    const int mjv = jvv - VV * ijv;
                    const float s = acc[v] * INV_TEMP;
                    const float e = __expf(s);
                    dacc[v] += (jvv != k) ? e : 0.0f;
                    pacc[v] += ((ik == ijv) && (mk != mjv)) ? s : 0.0f;
                }
            }
        }
    };

    // software-pipelined pair loop: prefetch B(kt+1) / B(kt+2) into the idle
    // buffer before computing the current one (loads complete in order, so
    // the s_wait before the WMMAs drains only to the prefetched count).
    V16 b0[4], b1[4];
    load_btile(z, 0, grp, r, b0);
    for (int kt = 0; kt < ntiles; kt += 2) {
        load_btile(z, kt + 1, grp, r, b1);
        compute_tile(kt, b0);
        if (kt + 2 < ntiles) load_btile(z, kt + 2, grp, r, b0);
        compute_tile(kt + 1, b1);
    }

    // reduce across the 16 lanes of each half-wave
#pragma unroll
    for (int v = 0; v < 8; ++v) {
#pragma unroll
        for (int off = 8; off >= 1; off >>= 1) {
            dacc[v] += __shfl_xor(dacc[v], off, 32);
            pacc[v] += __shfl_xor(pacc[v], off, 32);
        }
    }
    if (r == 0) {
#pragma unroll
        for (int v = 0; v < 8; ++v) {
            const int row = (jt << 4) + v + (grp << 3);
            denomO[row]  = dacc[v];
            possimO[row] = pacc[v];
        }
    }
}

// =====================================================================
// Kernel 4: single-block finalize (deterministic tree reductions)
// =====================================================================
__device__ float block_sum256(float v, float* sd) {
    const int t = threadIdx.x;
    sd[t] = v; __syncthreads();
#pragma unroll
    for (int s = 128; s > 0; s >>= 1) {
        if (t < s) sd[t] += sd[t + s];
        __syncthreads();
    }
    const float r = sd[0]; __syncthreads();
    return r;
}

__global__ void finalize_kernel(const float* __restrict__ supRows,
                                const float* __restrict__ unsRows,
                                const float* __restrict__ part,
                                float* __restrict__ out) {
    __shared__ float sd[256];
    const int t = threadIdx.x;

    // sup: cnt analytically = 1535 (label 1, rows < 1536) else 3071; valid = 1
    float a = 0.f;
    for (int r = t; r < KS; r += 256) {
        const float cnt = (r < NPOS * VV) ? CNT_POS : CNT_NEG;
        a += (cnt * logf(supRows[r] + 1e-12f) - supRows[KS + r]) / cnt;
    }
    const float supL = block_sum256(a, sd) / (float)KS;

    // unsup: cnt = 2 for every row; valid = 1
    a = 0.f;
    for (int r = t; r < KU; r += 256)
        a += (2.0f * logf(unsRows[r] + 1e-12f) - unsRows[KU + r]) * 0.5f;
    const float unsL = block_sum256(a, sd) / (float)KU;

    float q[5];
#pragma unroll
    for (int qi = 0; qi < 5; ++qi) {
        float s = 0.f;
        for (int p = t; p < BCE_BLOCKS; p += 256) s += part[p * 5 + qi];
        q[qi] = block_sum256(s, sd);
    }

    if (t == 0) {
        const float msum  = fmaxf(q[0], 1.0f);
        const float mainL = q[1] / msum;
        const float viewL = (q[2] + q[3] + q[4]) / (3.0f * msum);
        const float total = mainL + viewL + supL + L_UNSUP * unsL;
        out[0] = total; out[1] = mainL; out[2] = viewL; out[3] = supL; out[4] = unsL;
    }
}

// =====================================================================
// host launcher
// =====================================================================
extern "C" void kernel_launch(void* const* d_in, const int* in_sizes, int n_in,
                              void* d_out, int out_size, void* d_ws, size_t ws_size,
                              hipStream_t stream) {
    (void)in_sizes; (void)n_in; (void)out_size; (void)ws_size;

    const float*   fused  = (const float*)d_in[0];
    const float*   views  = (const float*)d_in[1];
    const float*   proj   = (const float*)d_in[2];
    const float*   labels = (const float*)d_in[3];
    const uint8_t* mask   = (const uint8_t*)d_in[4];
    const int*     posIdx = (const int*)d_in[5];
    const int*     negIdx = (const int*)d_in[6];
    const int*     unlIdx = (const int*)d_in[7];
    float*         out    = (float*)d_out;

    char* ws = (char*)d_ws;
    const size_t zsB = (size_t)KS * DD * sizeof(_Float16);
    const size_t zuB = (size_t)KU * DD * sizeof(_Float16);
    _Float16* zs      = (_Float16*)ws;
    _Float16* zu      = (_Float16*)(ws + zsB);
    float*    supRows = (float*)(ws + zsB + zuB);            // 2*KS floats (denom, possim)
    float*    unsRows = supRows + 2 * KS;                    // 2*KU floats
    float*    part    = unsRows + 2 * KU;                    // 64*5 floats

    bce_partials<<<BCE_BLOCKS, 256, 0, stream>>>(fused, views, labels, mask, part);

    gather_normalize<<<KS / 8, 256, 0, stream>>>(proj, posIdx, negIdx, NPOS, KS, zs);
    gather_normalize<<<KU / 8, 256, 0, stream>>>(proj, unlIdx, unlIdx, UU, KU, zu);

    gram_rowstats<0><<<KS / 16, 32, 0, stream>>>(zs, KS, supRows, supRows + KS);
    gram_rowstats<1><<<KU / 16, 32, 0, stream>>>(zu, KU, unsRows, unsRows + KU);

    finalize_kernel<<<1, 256, 0, stream>>>(supRows, unsRows, part, out);
}
